// GraphTransformerLinkPredictor_56951266345605
// MI455X (gfx1250) — matrix operand order repaired and checked by reading, not compiled, not run on hardware
//
#include <hip/hip_runtime.h>
#include <math.h>
#include <stdint.h>

typedef __bf16 bf16;
typedef __attribute__((ext_vector_type(16))) __bf16 bf16x16;
typedef __attribute__((ext_vector_type(8)))  __bf16 bf16x8;
typedef __attribute__((ext_vector_type(8)))  float  f32x8;

#define NN      3072
#define NE      147456
#define NP      100000
#define INCH    128
#define HID     256
#define HEADS   4
#define DH      64
#define WL      16
#define RWSE    16
#define KCAT    144   /* INCH + RWSE */
#define LAYERS  2

static __device__ __forceinline__ bf16 f2bf(float f) { return (bf16)f; }

// ---------------------------------------------------------------- utilities
__global__ void k_zero_f32(float* __restrict__ p, int n) {
  int i = blockIdx.x * blockDim.x + threadIdx.x;
  if (i < n) p[i] = 0.0f;
}

__global__ void k_scatter_edges(const int* __restrict__ ei, float* __restrict__ adj) {
  int e = blockIdx.x * blockDim.x + threadIdx.x;
  if (e < NE) {
    int s = ei[e];
    int t = ei[NE + e];
    atomicAdd(&adj[(size_t)s * NN + t], 1.0f);
  }
}

__global__ __launch_bounds__(256)
void k_rownorm(const float* __restrict__ adj, bf16* __restrict__ rw,
               float* __restrict__ diag0) {
  __shared__ float red[256];
  int r = blockIdx.x;
  int t = threadIdx.x;
  const float* row = adj + (size_t)r * NN;
  float s = 0.0f;
  for (int j = t; j < NN; j += 256) s += row[j];
  red[t] = s; __syncthreads();
  for (int w = 128; w > 0; w >>= 1) {
    if (t < w) red[t] += red[t + w];
    __syncthreads();
  }
  float inv = 1.0f / fmaxf(red[0], 1.0f);
  bf16* orow = rw + (size_t)r * NN;
  for (int j = t; j < NN; j += 256) {
    float v = row[j] * inv;
    orow[j] = f2bf(v);
    if (j == r) diag0[r] = v;
  }
}

// one-time bf16 transpose: out[n][k] = in[k][n]  (makes B-tiles contiguous)
__global__ __launch_bounds__(256)
void k_transpose_bf16(const bf16* __restrict__ in, bf16* __restrict__ out) {
  __shared__ bf16 tile[32][33];
  int bx = blockIdx.x * 32;   // input col block
  int by = blockIdx.y * 32;   // input row block
  int tx = threadIdx.x & 31, ty = threadIdx.x >> 5;  // 32 x 8
#pragma unroll
  for (int i = 0; i < 32; i += 8)
    tile[ty + i][tx] = in[(size_t)(by + ty + i) * NN + bx + tx];
  __syncthreads();
#pragma unroll
  for (int i = 0; i < 32; i += 8)
    out[(size_t)(bx + ty + i) * NN + by + tx] = tile[tx][ty + i];
}

// --------------------------- CDNA5 async global->LDS copy (ASYNCcnt-tracked)
// copies global data at vaddr+offset into LDS at vdst+offset  (16B per lane per op)
__device__ __forceinline__ void async_cp32B(unsigned lds_off, const bf16* g) {
  asm volatile("global_load_async_to_lds_b128 %0, %1, off"
               :: "v"(lds_off), "v"(g) : "memory");
  asm volatile("global_load_async_to_lds_b128 %0, %1, off offset:16"
               :: "v"(lds_off), "v"(g) : "memory");
}

// -------------------------------------------------- shared WMMA tile engine
// As: 128 rows x 32 K (row-major); Bt: 128 cols x 32 K (N-major, transposed B)
__device__ __forceinline__ void wmma_tile_accum(const bf16 (*As)[32], const bf16 (*Bt)[32],
                                                int wm, int wn, int lh, int hi,
                                                f32x8 acc[2][4]) {
#pragma unroll
  for (int sm = 0; sm < 2; ++sm) {
    int ar = wm * 32 + sm * 16 + lh;
    // wave32 A layout: lanes 0-15 hold K {0-7,16-23}; lanes 16-31 K {8-15,24-31}
    bf16x8 lo = *(const bf16x8*)&As[ar][hi * 8];
    bf16x8 hv = *(const bf16x8*)&As[ar][16 + hi * 8];
    bf16x16 af;
#pragma unroll
    for (int i = 0; i < 8; ++i) { af[i] = lo[i]; af[i + 8] = hv[i]; }
#pragma unroll
    for (int sn = 0; sn < 4; ++sn) {
      int br = wn * 64 + sn * 16 + lh;
      // wave32 B layout: lane = N, lanes 0-15 K {0-7,16-23}; lanes 16-31 K {8-15,24-31}
      bf16x8 blo = *(const bf16x8*)&Bt[br][hi * 8];
      bf16x8 bhv = *(const bf16x8*)&Bt[br][16 + hi * 8];
      bf16x16 bfv;
#pragma unroll
      for (int i = 0; i < 8; ++i) { bfv[i] = blo[i]; bfv[i + 8] = bhv[i]; }
      acc[sm][sn] = __builtin_amdgcn_wmma_f32_16x16x32_bf16(
          false, af, false, bfv, (short)0, acc[sm][sn], false, false);
    }
  }
}

// ------------------- bf16 x bf16 power GEMM, async double-buffered (+ diag)
__global__ __launch_bounds__(256)
void k_power_gemm(const bf16* __restrict__ A, const bf16* __restrict__ BT,
                  bf16* __restrict__ C, float* __restrict__ diag) {
  __shared__ bf16 As[2][128][32];
  __shared__ bf16 Bt[2][128][32];
  int tid = threadIdx.x;
  int m0 = blockIdx.x * 128;
  int n0 = blockIdx.y * 128;
  int lane = tid & 31, wave = tid >> 5;
  int wm = wave & 3, wn = wave >> 2;
  int lh = lane & 15, hi = lane >> 4;

  f32x8 zv = {};
  f32x8 acc[2][4];
#pragma unroll
  for (int a = 0; a < 2; ++a)
#pragma unroll
    for (int b = 0; b < 4; ++b) acc[a][b] = zv;

  int arow = tid >> 1, aoff = (tid & 1) * 16;   // 256 thr cover 128x32 bf16 tile

  auto issue_tile = [&](int buf, int kt) {
    const bf16* ap = A  + (size_t)(m0 + arow) * NN + kt + aoff;
    const bf16* bp = BT + (size_t)(n0 + arow) * NN + kt + aoff;
    unsigned lA = (unsigned)(uintptr_t)&As[buf][arow][aoff];  // flat[31:0] == LDS addr
    unsigned lB = (unsigned)(uintptr_t)&Bt[buf][arow][aoff];
    async_cp32B(lA, ap);   // 2 async ops
    async_cp32B(lB, bp);   // 2 async ops
  };

  issue_tile(0, 0);
  int buf = 0;
  for (int kt = 0; kt < NN; kt += 32) {
    if (kt + 32 < NN) {
      issue_tile(buf ^ 1, kt + 32);                       // 8 ops in flight
      asm volatile("s_wait_asynccnt 4" ::: "memory");     // current tile landed
    } else {
      asm volatile("s_wait_asynccnt 0" ::: "memory");
    }
    __syncthreads();
    wmma_tile_accum(As[buf], Bt[buf], wm, wn, lh, hi, acc);
    __syncthreads();
    buf ^= 1;
  }

#pragma unroll
  for (int sm = 0; sm < 2; ++sm)
#pragma unroll
    for (int sn = 0; sn < 4; ++sn) {
      int gn = n0 + wn * 64 + sn * 16 + lh;
#pragma unroll
      for (int r = 0; r < 8; ++r) {
        int gm = m0 + wm * 32 + sm * 16 + r + hi * 8;
        float v = acc[sm][sn][r];
        C[(size_t)gm * NN + gn] = f2bf(v);
        if (gm == gn) diag[gm] = v;   // random-walk return probability
      }
    }
}

// ------------------------------------ f32 GEMM (bf16 staged) with bias, K pad
__global__ __launch_bounds__(256)
void k_gemm_f32(const float* __restrict__ A, const float* __restrict__ B,
                const float* __restrict__ bias, float* __restrict__ C,
                int Ncols, int K) {
  __shared__ bf16 As[128][32];
  __shared__ bf16 Bt[128][32];
  int tid = threadIdx.x;
  int m0 = blockIdx.x * 128;
  int n0 = blockIdx.y * 128;
  int lane = tid & 31, wave = tid >> 5;
  int wm = wave & 3, wn = wave >> 2;
  int lh = lane & 15, hi = lane >> 4;

  f32x8 zv = {};
  f32x8 acc[2][4];
#pragma unroll
  for (int a = 0; a < 2; ++a)
#pragma unroll
    for (int b = 0; b < 4; ++b) acc[a][b] = zv;

  int arow = tid >> 1, aoff = (tid & 1) * 16;
  for (int kt = 0; kt < K; kt += 32) {
    if (kt + aoff < K) {   // K is a multiple of 16 -> chunk-level guard
      const float* ap = A + (size_t)(m0 + arow) * K + kt + aoff;
#pragma unroll
      for (int i = 0; i < 16; ++i) As[arow][aoff + i] = f2bf(ap[i]);
    } else {
#pragma unroll
      for (int i = 0; i < 16; ++i) As[arow][aoff + i] = f2bf(0.0f);
    }
#pragma unroll
    for (int i = 0; i < 16; ++i) {
      int lin = tid + 256 * i;
      int kk = lin >> 7, nc = lin & 127;
      float v = (kt + kk < K) ? B[(size_t)(kt + kk) * Ncols + n0 + nc] : 0.0f;
      Bt[nc][kk] = f2bf(v);
    }
    __syncthreads();
    wmma_tile_accum(As, Bt, wm, wn, lh, hi, acc);
    __syncthreads();
  }

#pragma unroll
  for (int sm = 0; sm < 2; ++sm)
#pragma unroll
    for (int sn = 0; sn < 4; ++sn) {
      int gn = n0 + wn * 64 + sn * 16 + lh;
      float bb = bias[gn];
#pragma unroll
      for (int r = 0; r < 8; ++r) {
        int gm = m0 + wm * 32 + sm * 16 + r + hi * 8;
        C[(size_t)gm * Ncols + gn] = acc[sm][sn][r] + bb;
      }
    }
}

// ------------------------------------------------------------ RWSE features
__global__ void k_build_xpe(const float* __restrict__ x, const float* __restrict__ diag,
                            const float* __restrict__ rwse_w, const float* __restrict__ rwse_b,
                            float* __restrict__ xpe) {
  int n = blockIdx.x, c = threadIdx.x;
  if (c < INCH) {
    xpe[(size_t)n * KCAT + c] = x[(size_t)n * INCH + c];
  } else if (c < KCAT) {
    int d = c - INCH;
    float a = rwse_b[d];
#pragma unroll
    for (int t = 0; t < WL; ++t) a += diag[t * NN + n] * rwse_w[t * RWSE + d];
    xpe[(size_t)n * KCAT + c] = a;
  }
}

// --------------------------------------------------------- edge attention
__device__ __forceinline__ unsigned f_enc(float f) {
  unsigned u = __float_as_uint(f);
  return (u & 0x80000000u) ? ~u : (u | 0x80000000u);
}
__device__ __forceinline__ float f_dec(unsigned u) {
  return __uint_as_float((u & 0x80000000u) ? (u & 0x7FFFFFFFu) : ~u);
}

__global__ void k_init_m_denom(unsigned* __restrict__ menc, float* __restrict__ denom) {
  int i = blockIdx.x * blockDim.x + threadIdx.x;
  if (i < NN * HEADS) { menc[i] = 0u; denom[i] = 0.0f; }
}

__global__ void k_scores(const int* __restrict__ ei, const float* __restrict__ q,
                         const float* __restrict__ kmat, float* __restrict__ sc,
                         unsigned* __restrict__ menc) {
  int i = blockIdx.x * blockDim.x + threadIdx.x;
  if (i >= NE * HEADS) return;
  int e = i >> 2, hd = i & 3;
  int s = ei[e], t = ei[NE + e];
  const float4* qp = (const float4*)(q + (size_t)t * HID + hd * DH);
  const float4* kp = (const float4*)(kmat + (size_t)s * HID + hd * DH);
  float acc = 0.0f;
#pragma unroll
  for (int j = 0; j < DH / 4; ++j) {
    float4 a = qp[j], b = kp[j];
    acc += a.x * b.x + a.y * b.y + a.z * b.z + a.w * b.w;
  }
  acc *= 0.125f;  // 1/sqrt(DH)
  sc[i] = acc;
  atomicMax(&menc[t * HEADS + hd], f_enc(acc));
}

__global__ void k_decode_m(unsigned* __restrict__ m) {
  int i = blockIdx.x * blockDim.x + threadIdx.x;
  if (i < NN * HEADS) {
    unsigned u = m[i];
    ((float*)m)[i] = (u == 0u) ? 0.0f : f_dec(u);  // empty segment -> 0 like ref
  }
}

__global__ void k_exp_denom(const int* __restrict__ ei, float* __restrict__ sc,
                            const float* __restrict__ m, float* __restrict__ denom) {
  int i = blockIdx.x * blockDim.x + threadIdx.x;
  if (i >= NE * HEADS) return;
  int e = i >> 2, hd = i & 3;
  int t = ei[NE + e];
  float ev = __expf(sc[i] - m[t * HEADS + hd]);
  sc[i] = ev;
  atomicAdd(&denom[t * HEADS + hd], ev);
}

__global__ void k_aggregate(const int* __restrict__ ei, const float* __restrict__ sc,
                            const float* __restrict__ denom, const float* __restrict__ v,
                            float* __restrict__ agg) {
  int i = blockIdx.x * blockDim.x + threadIdx.x;
  if (i >= NE * HID) return;
  int e = i >> 8, c = i & 255;
  int hd = c >> 6;
  int s = ei[e], t = ei[NE + e];
  float alpha = sc[e * HEADS + hd] / fmaxf(denom[t * HEADS + hd], 1e-16f);
  atomicAdd(&agg[(size_t)t * HID + c], alpha * v[(size_t)s * HID + c]);
}

__global__ __launch_bounds__(256)
void k_residual_ln(float* __restrict__ h, const float* __restrict__ agg,
                   const float* __restrict__ skip, const float* __restrict__ g,
                   const float* __restrict__ b) {
  __shared__ float red[256];
  __shared__ float red2[256];
  int n = blockIdx.x, c = threadIdx.x;
  size_t idx = (size_t)n * HID + c;
  float val = h[idx] + agg[idx] + skip[idx];
  red[c] = val; red2[c] = val * val; __syncthreads();
  for (int w = 128; w > 0; w >>= 1) {
    if (c < w) { red[c] += red[c + w]; red2[c] += red2[c + w]; }
    __syncthreads();
  }
  float mu = red[0] * (1.0f / HID);
  float var = red2[0] * (1.0f / HID) - mu * mu;
  float rs = rsqrtf(fmaxf(var, 0.0f) + 1e-5f);
  float o = (val - mu) * rs * g[c] + b[c];
  h[idx] = fmaxf(o, 0.0f);
}

__global__ void k_link_decode(const int* __restrict__ src, const int* __restrict__ dst,
                              const float* __restrict__ h, float* __restrict__ out) {
  int p = blockIdx.x * blockDim.x + threadIdx.x;
  if (p >= NP) return;
  const float4* a = (const float4*)(h + (size_t)src[p] * HID);
  const float4* b = (const float4*)(h + (size_t)dst[p] * HID);
  float acc = 0.0f;
#pragma unroll 8
  for (int j = 0; j < HID / 4; ++j) {
    float4 x = a[j], y = b[j];
    acc += x.x * y.x + x.y * y.y + x.z * y.z + x.w * y.w;
  }
  out[p] = 1.0f / (1.0f + __expf(-acc));
}

// ------------------------------------------------------------------- driver
extern "C" void kernel_launch(void* const* d_in, const int* in_sizes, int n_in,
                              void* d_out, int out_size, void* d_ws, size_t ws_size,
                              hipStream_t stream) {
  const float* x      = (const float*)d_in[0];
  const int*   ei     = (const int*)d_in[1];
  const int*   src    = (const int*)d_in[2];
  const int*   dst    = (const int*)d_in[3];
  const float* rwse_w = (const float*)d_in[4];
  const float* rwse_b = (const float*)d_in[5];
  const float* proj_w = (const float*)d_in[6];
  const float* proj_b = (const float*)d_in[7];
  const float* Wq = (const float*)d_in[8];
  const float* bq = (const float*)d_in[9];
  const float* Wk = (const float*)d_in[10];
  const float* bk = (const float*)d_in[11];
  const float* Wv = (const float*)d_in[12];
  const float* bv = (const float*)d_in[13];
  const float* Wsk = (const float*)d_in[14];
  const float* bsk = (const float*)d_in[15];
  const float* ln_g = (const float*)d_in[16];
  const float* ln_b = (const float*)d_in[17];

  // --- workspace layout (region0 = adjacency, reused by transformer bufs) ---
  char* ws = (char*)d_ws;
  size_t off = 0;
  float* adj = (float*)(ws + off);  off += (size_t)NN * NN * 4;   // 37.75 MB
  bf16*  rw  = (bf16*)(ws + off);   off += (size_t)NN * NN * 2;
  bf16*  rwT = (bf16*)(ws + off);   off += (size_t)NN * NN * 2;   // transposed B
  bf16*  pA  = (bf16*)(ws + off);   off += (size_t)NN * NN * 2;
  bf16*  pB  = (bf16*)(ws + off);   off += (size_t)NN * NN * 2;
  float* diag = (float*)(ws + off); off += (size_t)WL * NN * 4;

  char* r0 = (char*)d_ws;           // overlays adj once power chain is done
  float*    xpe   = (float*)r0;     r0 += (size_t)NN * KCAT * 4;
  float*    h     = (float*)r0;     r0 += (size_t)NN * HID * 4;
  float*    qb    = (float*)r0;     r0 += (size_t)NN * HID * 4;
  float*    kb    = (float*)r0;     r0 += (size_t)NN * HID * 4;
  float*    vb    = (float*)r0;     r0 += (size_t)NN * HID * 4;
  float*    skip  = (float*)r0;     r0 += (size_t)NN * HID * 4;
  float*    sc    = (float*)r0;     r0 += (size_t)NE * HEADS * 4;
  unsigned* menc  = (unsigned*)r0;  r0 += (size_t)NN * HEADS * 4;
  float*    denom = (float*)r0;     r0 += (size_t)NN * HEADS * 4;
  float*    agg   = (float*)r0;     r0 += (size_t)NN * HID * 4;

  const int tz = 256;

  // 1) adjacency build + row-normalize (rw bf16, diag[0] = diag(rw))
  k_zero_f32<<<(NN * NN + tz - 1) / tz, tz, 0, stream>>>(adj, NN * NN);
  k_scatter_edges<<<(NE + tz - 1) / tz, tz, 0, stream>>>(ei, adj);
  k_rownorm<<<NN, 256, 0, stream>>>(adj, rw, diag);
  k_transpose_bf16<<<dim3(NN / 32, NN / 32), 256, 0, stream>>>(rw, rwT);

  // 2) random-walk powers: diag[t] = diag(rw^(t+1)) — the 0.87 TFLOP WMMA chain
  dim3 pgrid(NN / 128, NN / 128);
  const bf16* curA = rw;
  for (int t = 1; t < WL; ++t) {
    bf16* curC = (t & 1) ? pA : pB;
    k_power_gemm<<<pgrid, 256, 0, stream>>>(curA, rwT, curC, diag + t * NN);
    curA = curC;
  }

  // 3) [x | pe] projection
  k_build_xpe<<<NN, 160, 0, stream>>>(x, diag, rwse_w, rwse_b, xpe);
  dim3 ggrid(NN / 128, HID / 128);
  k_gemm_f32<<<ggrid, 256, 0, stream>>>(xpe, proj_w, proj_b, h, HID, KCAT);

  // 4) transformer layers
  for (int l = 0; l < LAYERS; ++l) {
    size_t wofs = (size_t)l * HID * HID;
    k_gemm_f32<<<ggrid, 256, 0, stream>>>(h, Wq + wofs, bq + l * HID, qb, HID, HID);
    k_gemm_f32<<<ggrid, 256, 0, stream>>>(h, Wk + wofs, bk + l * HID, kb, HID, HID);
    k_gemm_f32<<<ggrid, 256, 0, stream>>>(h, Wv + wofs, bv + l * HID, vb, HID, HID);
    k_gemm_f32<<<ggrid, 256, 0, stream>>>(h, Wsk + wofs, bsk + l * HID, skip, HID, HID);

    k_init_m_denom<<<(NN * HEADS + tz - 1) / tz, tz, 0, stream>>>(menc, denom);
    k_zero_f32<<<(NN * HID + tz - 1) / tz, tz, 0, stream>>>(agg, NN * HID);

    k_scores<<<(NE * HEADS + tz - 1) / tz, tz, 0, stream>>>(ei, qb, kb, sc, menc);
    k_decode_m<<<(NN * HEADS + tz - 1) / tz, tz, 0, stream>>>(menc);
    k_exp_denom<<<(NE * HEADS + tz - 1) / tz, tz, 0, stream>>>(ei, sc, (const float*)menc, denom);
    k_aggregate<<<(NE * HID + tz - 1) / tz, tz, 0, stream>>>(ei, sc, denom, vb, agg);
    k_residual_ln<<<NN, HID, 0, stream>>>(h, agg, skip, ln_g + l * HID, ln_b + l * HID);
  }

  // 5) link decode
  k_link_decode<<<(NP + tz - 1) / tz, tz, 0, stream>>>(src, dst, h, (float*)d_out);
}
